// RNN_49684181680264
// MI455X (gfx1250) — matrix operand order repaired and checked by reading, not compiled, not run on hardware
//
#include <hip/hip_runtime.h>

typedef __attribute__((ext_vector_type(16))) _Float16 v16h;
typedef __attribute__((ext_vector_type(2)))  _Float16 v2h;
typedef __attribute__((ext_vector_type(8)))  float    v8f;
typedef __attribute__((ext_vector_type(8)))  unsigned v8u;
typedef __attribute__((ext_vector_type(2)))  unsigned v2u;

// Problem constants (match reference).
constexpr int Bn = 16384;
constexpr int Tn = 1024;
constexpr int Hn = 32;

constexpr int WAVES = 8;   // waves per block; each wave owns 16 batch rows
constexpr int TC    = 64;  // timesteps staged per LDS chunk

// Branch-free tanh: CDNA5 hardware v_tanh_f32.
__device__ __forceinline__ float fast_tanh(float v) {
#if __has_builtin(__builtin_amdgcn_tanhf)
    return __builtin_amdgcn_tanhf(v);
#else
    const float e = __builtin_amdgcn_exp2f(v * 2.8853900817779268f);
    const float r = __builtin_amdgcn_rcpf(e + 1.0f);
    return __builtin_fmaf(-2.0f, r, 1.0f);
#endif
}

__global__ __launch_bounds__(WAVES * 32)
void rnn_wmma_kernel(const float* __restrict__ x,
                     const float* __restrict__ Wxh,
                     const float* __restrict__ b_xh,
                     const float* __restrict__ Whh,
                     const float* __restrict__ b_hh,
                     const float* __restrict__ Wout,
                     const float* __restrict__ b_out,
                     float* __restrict__ out)
{
    // Per-wave private staging region: xs[wave][t][batch_col], f32.
    __shared__ float xs[WAVES][TC][16];

    const int lane = threadIdx.x & 31;
    const int wave = threadIdx.x >> 5;
    const int n    = lane & 15;   // batch column within tile (WMMA N / lane)
    const int hi   = lane >> 4;   // lane half (selects K/M sub-range)
    const int b0   = (blockIdx.x * WAVES + wave) * 16;

    // ---- Loop-invariant per-lane constants -------------------------------
    // D-tile layout: lane L, VGPR k holds (M = k + 8*hi, N = L%16).
    float cb0[8], cb1[8], wx0[8], wx1[8], wo0[8], wo1[8];
#pragma unroll
    for (int k = 0; k < 8; ++k) {
        const int j0 = k + 8 * hi;
        const int j1 = j0 + 16;
        cb0[k] = b_hh[j0] + b_xh[j0];
        cb1[k] = b_hh[j1] + b_xh[j1];
        wx0[k] = Wxh[j0];
        wx1[k] = Wxh[j1];
        wo0[k] = Wout[j0];
        wo1[k] = Wout[j1];
    }

    // ---- Stationary A fragments: A = Whh^T (M=hidden_out, K=hidden_in) ---
    // 16-bit A 16x32 layout (contiguous K): lane L, VGPR v -> K=2v,2v+1;
    // hi lanes: +16 on K.  Element (M=jo,K=ji) = Whh[ji*H+jo].
    v16h A0, A1;
#pragma unroll
    for (int v = 0; v < 8; ++v) {
        const int ji = 2 * v + 16 * hi;
        A0[2 * v]     = (_Float16)Whh[ji * Hn + n];
        A0[2 * v + 1] = (_Float16)Whh[(ji + 1) * Hn + n];
        A1[2 * v]     = (_Float16)Whh[ji * Hn + 16 + n];
        A1[2 * v + 1] = (_Float16)Whh[(ji + 1) * Hn + 16 + n];
    }

    // Recurrent state: B fragment h^T (32 K x 16 batch), starts at zero.
    v8u zz = {};
    v16h hB = __builtin_bit_cast(v16h, zz);
    float hf0[8] = {}, hf1[8] = {};   // last h (f32, hidden-major) for epilogue

    for (int tc = 0; tc < Tn; tc += TC) {
        // ---- Stage x chunk: coalesced float4 loads along T, transpose to LDS.
#pragma unroll
        for (int it = 0; it < 8; ++it) {
            const int r = it * 2 + hi;
            const float4 v4 =
                reinterpret_cast<const float4*>(x + (size_t)(b0 + r) * Tn + tc)[n];
            xs[wave][n * 4 + 0][r] = v4.x;
            xs[wave][n * 4 + 1][r] = v4.y;
            xs[wave][n * 4 + 2][r] = v4.z;
            xs[wave][n * 4 + 3][r] = v4.w;
        }
        // No barrier: region is private to this wave; DS ops are in-order per wave.

        // ---- Recurrent inner loop -----------------------------------------
#pragma unroll 4
        for (int tt = 0; tt < TC; ++tt) {
            const float xt = xs[wave][tt][n];

            // C = combined bias + x_t * Wxh, fed straight into WMMA accumulator.
            v8f c0, c1;
#pragma unroll
            for (int k = 0; k < 8; ++k) {
                c0[k] = fmaf(xt, wx0[k], cb0[k]);
                c1[k] = fmaf(xt, wx1[k], cb1[k]);
            }

            // D = Whh^T @ h^T + C   (two 16x16 tiles: hidden 0-15 / 16-31)
            v8f d0 = __builtin_amdgcn_wmma_f32_16x16x32_f16(
                false, A0, false, hB, (short)0, c0, false, false);
            v8f d1 = __builtin_amdgcn_wmma_f32_16x16x32_f16(
                false, A1, false, hB, (short)0, c1, false, false);

            // Nonlinearity: one v_tanh_f32 per element, branch-free.
#pragma unroll
            for (int k = 0; k < 8; ++k) {
                hf0[k] = fast_tanh(d0[k]);
                hf1[k] = fast_tanh(d1[k]);
            }

            // Repack D (M=hidden across VGPRs, lane=batch) into the B fragment
            // (lane=batch, K=hidden packed f16 pairs). Only the lane^16
            // half-exchange is needed. v_permlane16_swap_b32 does it as a pure
            // VALU op (no LDS pipe / DScnt on the serial critical path) and
            // yields BOTH combined registers at once.
            unsigned p0[4], p1[4];
#pragma unroll
            for (int j = 0; j < 4; ++j) {
                p0[j] = __builtin_bit_cast(
                    unsigned, __builtin_amdgcn_cvt_pkrtz(hf0[2 * j], hf0[2 * j + 1]));
                p1[j] = __builtin_bit_cast(
                    unsigned, __builtin_amdgcn_cvt_pkrtz(hf1[2 * j], hf1[2 * j + 1]));
            }
            v8u nb;
#pragma unroll
            for (int j = 0; j < 4; ++j) {
#if __has_builtin(__builtin_amdgcn_permlane16_swap)
                // Exchange 16-lane halves of (p0, p1):
                //   one result = {lo: own tile0, hi: partner tile1}   -> nb[j]
                //   other      = {lo: partner tile0, hi: own tile1}   -> nb[4+j]
                const v2u r = __builtin_bit_cast(
                    v2u, __builtin_amdgcn_permlane16_swap(p0[j], p1[j], false, false));
                nb[j]     = r.x;
                nb[4 + j] = r.y;
#else
                const unsigned src = hi ? p0[j] : p1[j];
                const unsigned q =
                    (unsigned)__builtin_amdgcn_ds_swizzle((int)src, 0x401F); // SWAPX16
                nb[j]     = hi ? q : p0[j];
                nb[4 + j] = hi ? p1[j] : q;
#endif
            }
            hB = __builtin_bit_cast(v16h, nb);
        }
    }

    // ---- Output projection: out[b] = sum_j h[b][j]*Wout[j] + b_out -------
    float s = 0.f;
#pragma unroll
    for (int k = 0; k < 8; ++k)
        s = fmaf(hf0[k], wo0[k], fmaf(hf1[k], wo1[k], s));
    // Partner lane (lane^16) holds the other 16 hidden partials for batch n.
    const float sp = __builtin_bit_cast(
        float, __builtin_amdgcn_ds_swizzle(__builtin_bit_cast(int, s), 0x401F));
    if (hi == 0)
        out[b0 + n] = s + sp + b_out[0];
}

extern "C" void kernel_launch(void* const* d_in, const int* in_sizes, int n_in,
                              void* d_out, int out_size, void* d_ws, size_t ws_size,
                              hipStream_t stream) {
    const float* x     = (const float*)d_in[0];
    const float* Wxh   = (const float*)d_in[1];
    const float* b_xh  = (const float*)d_in[2];
    const float* Whh   = (const float*)d_in[3];
    const float* b_hh  = (const float*)d_in[4];
    const float* Wout  = (const float*)d_in[5];
    const float* b_out = (const float*)d_in[6];
    float* out = (float*)d_out;

    const int blocks = Bn / (WAVES * 16);  // 16384 / 128 = 128 blocks
    rnn_wmma_kernel<<<blocks, WAVES * 32, 0, stream>>>(
        x, Wxh, b_xh, Whh, b_hh, Wout, b_out, out);
}
